// BinaryDenseNet_87024627351894
// MI455X (gfx1250) — 16-bit and 8-bit matrix operand order repaired and checked by reading, compiled, not run on hardware
//
#include <hip/hip_runtime.h>
#include <stdint.h>


typedef __attribute__((ext_vector_type(8)))  int      v8i;
typedef __attribute__((ext_vector_type(4)))  int      v4i;
typedef __attribute__((ext_vector_type(2)))  int      v2i;
typedef __attribute__((ext_vector_type(16))) _Float16 v16h;
typedef __attribute__((ext_vector_type(8)))  float    v8f;

__device__ __forceinline__ void async_b128(void* lds, const void* gaddr) {
  unsigned l = (unsigned)(uintptr_t)lds;
  asm volatile("global_load_async_to_lds_b128 %0, %1, off"
               :: "v"(l), "v"(gaddr) : "memory");
}
__device__ __forceinline__ void wait_async() {
  asm volatile("s_wait_asynccnt 0" ::: "memory");
}

__global__ __launch_bounds__(256) void k_bn_sign(
    const float* __restrict__ h, const float* __restrict__ g,
    const float* __restrict__ be, const float* __restrict__ mu,
    const float* __restrict__ va, int8_t* __restrict__ out,
    int M, int C, int Cstride, int Cpad)
{
  long long i = (long long)blockIdx.x * 256 + threadIdx.x;
  long long tot = (long long)M * Cpad;
  if (i >= tot) return;
  int c = (int)(i % Cpad);
  long long m = i / Cpad;
  int8_t v = 0;
  if (c < C) {
    float x = h[m * Cstride + c];
    float y = g[c] * (x - mu[c]) * rsqrtf(va[c] + 1e-5f) + be[c];
    v = (y >= 0.f) ? (int8_t)1 : (int8_t)-1;
  }
  out[i] = v;
}

__global__ __launch_bounds__(256) void k_bin_weights(
    const float* __restrict__ w, int8_t* __restrict__ out, int C, int Cpad)
{
  int i = blockIdx.x * 256 + threadIdx.x;
  int tot = 9 * 64 * Cpad;
  if (i >= tot) return;
  int k = i % Cpad;
  int r = i / Cpad;
  int n = r % 64;
  int j = r / 64;
  int8_t v = 0;
  if (k < C) v = (w[((size_t)j * C + k) * 64 + n] >= 0.f) ? (int8_t)1 : (int8_t)-1;
  out[i] = v;
}

__global__ __launch_bounds__(256) void k_binconv3x3_iu8(
    const int8_t* __restrict__ A, const int8_t* __restrict__ Wt,
    float* __restrict__ out, int Bn, int H, int W, int Cpad,
    int coutBase, int coutStride)
{
  __shared__ __attribute__((aligned(16))) int8_t As[2][128 * 64];
  __shared__ __attribute__((aligned(16))) int8_t Bs[2][64 * 64];

  const int tid  = threadIdx.x;
  const int lane = tid & 31;
  const int wid  = tid >> 5;
  const int wm   = wid >> 1;
  const int wn   = wid & 1;
  const int HW   = H * W;
  const int M    = Bn * HW;
  const int m0   = blockIdx.x * 128;

  v8i acc[2][2] = {};

  const int ar = tid >> 1;
  const int ao = (tid & 1) << 5;
  const int br = tid >> 2;
  const int bo = (tid & 3) << 4;

  const int am  = m0 + ar;
  const int ab  = am / HW;
  const int rem = am - ab * HW;
  const int ay  = rem / W;
  const int ax  = rem - ay * W;
  const bool mok = am < M;

  const int nc64    = Cpad >> 6;
  const int nChunks = 9 * nc64;

  auto stage = [&](int t, int buf) {
    const int j  = t / nc64;
    const int c0 = (t - j * nc64) << 6;
    const int ky = j / 3;
    const int kx = j - ky * 3;
    const int iy = ay + ky - 1;
    const int ix = ax + kx - 1;
    const bool ok = mok && iy >= 0 && iy < H && ix >= 0 && ix < W;
    int8_t* dA = &As[buf][ar * 64 + ao];
    if (ok) {
      const int8_t* src = A + ((size_t)(ab * H + iy) * W + ix) * Cpad + c0 + ao;
      async_b128(dA,      src);
      async_b128(dA + 16, src + 16);
    } else {
      v4i z = {};
      *(v4i*)(dA)      = z;
      *(v4i*)(dA + 16) = z;
    }
    async_b128(&Bs[buf][br * 64 + bo],
               Wt + (size_t)(j * 64 + br) * Cpad + c0 + bo);
  };

  stage(0, 0);
  for (int t = 0; t < nChunks; ++t) {
    const int buf = t & 1;
    wait_async();
    __syncthreads();
    if (t + 1 < nChunks) stage(t + 1, buf ^ 1);

    v8i af[2], bf[2];
    const int ahb = (lane < 16) ? 0 : 8;
#pragma unroll
    for (int mt = 0; mt < 2; ++mt) {
      const int row = wm * 32 + mt * 16 + (lane & 15);
      const int8_t* p = &As[buf][row * 64 + ahb];
#pragma unroll
      for (int g = 0; g < 4; ++g) {
        v2i tt = *(const v2i*)(p + g * 16);
        af[mt][2 * g]     = tt.x;
        af[mt][2 * g + 1] = tt.y;
      }
    }
    const int bhb = (lane < 16) ? 0 : 8;
#pragma unroll
    for (int nt = 0; nt < 2; ++nt) {
      const int col = wn * 32 + nt * 16 + (lane & 15);
      const int8_t* p = &Bs[buf][col * 64 + bhb];
#pragma unroll
      for (int g = 0; g < 4; ++g) {
        v2i tt = *(const v2i*)(p + g * 16);
        bf[nt][2 * g]     = tt.x;
        bf[nt][2 * g + 1] = tt.y;
      }
    }
#pragma unroll
    for (int mt = 0; mt < 2; ++mt)
#pragma unroll
      for (int nt = 0; nt < 2; ++nt)
        acc[mt][nt] = __builtin_amdgcn_wmma_i32_16x16x64_iu8(
            true, af[mt], true, bf[nt], acc[mt][nt], false, false);
  }

#pragma unroll
  for (int mt = 0; mt < 2; ++mt)
#pragma unroll
    for (int nt = 0; nt < 2; ++nt) {
      const int nc = coutBase + wn * 32 + nt * 16 + (lane & 15);
#pragma unroll
      for (int r = 0; r < 8; ++r) {
        const int mr = m0 + wm * 32 + mt * 16 + r + ((lane < 16) ? 0 : 8);
        if (mr < M) out[(size_t)mr * coutStride + nc] = (float)acc[mt][nt][r];
      }
    }
}

__global__ __launch_bounds__(256) void k_gemm_f16(
    const _Float16* __restrict__ A, const _Float16* __restrict__ Bt,
    const float* __restrict__ bias, float* __restrict__ out,
    int M, int N, int K, int opitch, int obase)
{
  __shared__ __attribute__((aligned(16))) _Float16 Ah[2][128 * 32];
  __shared__ __attribute__((aligned(16))) _Float16 Bh[2][32 * 32];

  const int tid  = threadIdx.x;
  const int lane = tid & 31;
  const int wid  = tid >> 5;
  const int wm   = wid >> 1;
  const int wn   = wid & 1;
  const int m0   = blockIdx.x * 128;
  const int n0   = blockIdx.y * 32;

  v8f acc[2] = {};

  const int arr = tid >> 1;
  const int akc = (tid & 1) * 16;
  const int brr = tid >> 2;
  const int bkc = (tid & 3) * 8;
  const bool aok = (m0 + arr) < M;
  const bool bok = (n0 + brr) < N;

  auto stage = [&](int k0, int buf) {
    _Float16* dA = &Ah[buf][arr * 32 + akc];
    if (aok) {
      const _Float16* src = A + (size_t)(m0 + arr) * K + k0 + akc;
      async_b128(dA,     src);
      async_b128(dA + 8, src + 8);
    } else {
      v4i z = {};
      *(v4i*)(dA)     = z;
      *(v4i*)(dA + 8) = z;
    }
    if (tid < 128) {
      _Float16* dB = &Bh[buf][brr * 32 + bkc];
      if (bok) {
        async_b128(dB, Bt + (size_t)(n0 + brr) * K + k0 + bkc);
      } else {
        v4i z = {};
        *(v4i*)(dB) = z;
      }
    }
  };

  const int nK = K >> 5;
  stage(0, 0);
  for (int kt = 0; kt < nK; ++kt) {
    const int buf = kt & 1;
    wait_async();
    __syncthreads();
    if (kt + 1 < nK) stage((kt + 1) << 5, buf ^ 1);

    union Frag { v16h h; v4i i[2]; };
    Frag bf;
    {
      const int col = wn * 16 + (lane & 15);
      const int kb  = (lane < 16) ? 0 : 16;
      bf.i[0] = *(const v4i*)(&Bh[buf][col * 32 + kb]);
      bf.i[1] = *(const v4i*)(&Bh[buf][col * 32 + kb + 8]);
    }
#pragma unroll
    for (int mt = 0; mt < 2; ++mt) {
      Frag af;
      const int row = wm * 32 + mt * 16 + (lane & 15);
      const int kb  = (lane < 16) ? 0 : 16;
      af.i[0] = *(const v4i*)(&Ah[buf][row * 32 + kb]);
      af.i[1] = *(const v4i*)(&Ah[buf][row * 32 + kb + 8]);
      acc[mt] = __builtin_amdgcn_wmma_f32_16x16x32_f16(
          false, af.h, false, bf.h, (short)0, acc[mt], false, false);
    }
  }

  const int nc = n0 + wn * 16 + (lane & 15);
  if (nc < N) {
    const float bv = bias ? bias[nc] : 0.f;
#pragma unroll
    for (int mt = 0; mt < 2; ++mt)
#pragma unroll
      for (int r = 0; r < 8; ++r) {
        const int mr = m0 + wm * 32 + mt * 16 + r + ((lane < 16) ? 0 : 8);
        if (mr < M) out[(size_t)mr * opitch + obase + nc] = acc[mt][r] + bv;
      }
  }
}

__global__ __launch_bounds__(256) void k_stem_conv(
    const float* __restrict__ x, const float* __restrict__ w,
    float* __restrict__ out, int Bn)
{
  int i = blockIdx.x * 256 + threadIdx.x;
  int tot = Bn * 112 * 112 * 64;
  if (i >= tot) return;
  int co = i & 63;
  int p = i >> 6;
  int ox = p % 112; p /= 112;
  int oy = p % 112;
  int b  = p / 112;
  float s = 0.f;
  for (int ky = 0; ky < 7; ++ky) {
    int iy = oy * 2 + ky - 2;
    if (iy < 0 || iy >= 224) continue;
    for (int kx = 0; kx < 7; ++kx) {
      int ix = ox * 2 + kx - 2;
      if (ix < 0 || ix >= 224) continue;
      const float* px = x + ((size_t)(b * 224 + iy) * 224 + ix) * 3;
      const float* pw = w + (size_t)(ky * 7 + kx) * 3 * 64 + co;
      s += px[0] * pw[0] + px[1] * pw[64] + px[2] * pw[128];
    }
  }
  out[i] = s;
}

__global__ __launch_bounds__(256) void k_bn_act(
    const float* __restrict__ in, const float* __restrict__ g,
    const float* __restrict__ be, const float* __restrict__ mu,
    const float* __restrict__ va, float* __restrict__ out,
    int M, int C, int ipitch, int opitch, int relu)
{
  long long i = (long long)blockIdx.x * 256 + threadIdx.x;
  if (i >= (long long)M * C) return;
  int c = (int)(i % C);
  long long m = i / C;
  float y = g[c] * (in[m * ipitch + c] - mu[c]) * rsqrtf(va[c] + 1e-5f) + be[c];
  if (relu) y = fmaxf(y, 0.f);
  out[m * opitch + c] = y;
}

__global__ __launch_bounds__(256) void k_maxpool(
    const float* __restrict__ in, float* __restrict__ out,
    int Bn, int Hin, int Win, int Hout, int Wout, int C,
    int ipitch, int opitch, int win, int stride, int relu)
{
  long long i = (long long)blockIdx.x * 256 + threadIdx.x;
  long long tot = (long long)Bn * Hout * Wout * C;
  if (i >= tot) return;
  int c = (int)(i % C);
  long long p = i / C;
  int ox = (int)(p % Wout); p /= Wout;
  int oy = (int)(p % Hout);
  int b  = (int)(p / Hout);
  float m = -3.0e38f;
  for (int wy = 0; wy < win; ++wy) {
    int iy = oy * stride + wy;
    if (iy >= Hin) continue;
    for (int wx = 0; wx < win; ++wx) {
      int ix = ox * stride + wx;
      if (ix >= Win) continue;
      m = fmaxf(m, in[((size_t)(b * Hin + iy) * Win + ix) * ipitch + c]);
    }
  }
  if (relu) m = fmaxf(m, 0.f);
  out[((size_t)(b * Hout + oy) * Wout + ox) * opitch + c] = m;
}

__global__ __launch_bounds__(256) void k_f32_to_f16(
    const float* __restrict__ in, _Float16* __restrict__ out,
    int M, int C, int ipitch)
{
  long long i = (long long)blockIdx.x * 256 + threadIdx.x;
  if (i >= (long long)M * C) return;
  int c = (int)(i % C);
  long long m = i / C;
  out[i] = (_Float16)in[m * ipitch + c];
}

__global__ __launch_bounds__(256) void k_w_to_f16t(
    const float* __restrict__ w, _Float16* __restrict__ out, int K, int N)
{
  long long i = (long long)blockIdx.x * 256 + threadIdx.x;
  if (i >= (long long)N * K) return;
  int k = (int)(i % K);
  int n = (int)(i / K);
  out[i] = (_Float16)w[(size_t)k * N + n];
}

__global__ __launch_bounds__(256) void k_gap(
    const float* __restrict__ in, float* __restrict__ out, int Bn, int HW, int C)
{
  int i = blockIdx.x * 256 + threadIdx.x;
  if (i >= Bn * C) return;
  int c = i % C;
  int b = i / C;
  float s = 0.f;
  for (int p = 0; p < HW; ++p) s += in[((size_t)b * HW + p) * C + c];
  out[i] = s / (float)HW;
}

__global__ __launch_bounds__(256) void k_softmax(
    const float* __restrict__ in, float* __restrict__ out, int N)
{
  __shared__ float red[256];
  const int row = blockIdx.x;
  const int tid = threadIdx.x;
  const float* r = in + (size_t)row * N;
  float mx = -3.0e38f;
  for (int c = tid; c < N; c += 256) mx = fmaxf(mx, r[c]);
  red[tid] = mx; __syncthreads();
  for (int s = 128; s > 0; s >>= 1) {
    if (tid < s) red[tid] = fmaxf(red[tid], red[tid + s]);
    __syncthreads();
  }
  mx = red[0]; __syncthreads();
  float sm = 0.f;
  for (int c = tid; c < N; c += 256) sm += __expf(r[c] - mx);
  red[tid] = sm; __syncthreads();
  for (int s = 128; s > 0; s >>= 1) {
    if (tid < s) red[tid] += red[tid + s];
    __syncthreads();
  }
  const float inv = 1.f / red[0];
  for (int c = tid; c < N; c += 256)
    out[(size_t)row * N + c] = __expf(r[c] - mx) * inv;
}

static inline unsigned cdiv(long long a, long long b) { return (unsigned)((a + b - 1) / b); }

extern "C" void kernel_launch(void* const* d_in, const int* in_sizes, int n_in,
                              void* d_out, int out_size, void* d_ws, size_t ws_size,
                              hipStream_t stream)
{
  (void)in_sizes; (void)n_in; (void)out_size; (void)ws_size;

  static const int layers[4] = {6, 6, 6, 5};
  static const int Hs[4]     = {56, 28, 14, 7};
  static const int Cstart[4] = {64, 160, 192, 256};
  static const int Cend[4]   = {448, 544, 576, 576};
  static const int tcout[3]  = {160, 192, 256};
  const int Bn = 8;

  int idx = 0;
  const float* X      = (const float*)d_in[idx++];
  const float* stem_w = (const float*)d_in[idx++];
  const float* stem_bn[4];
  for (int j = 0; j < 4; ++j) stem_bn[j] = (const float*)d_in[idx++];
  const float* blk_bn[4][6][4];
  const float* blk_w[4][6];
  for (int b = 0; b < 4; ++b)
    for (int l = 0; l < layers[b]; ++l) {
      for (int j = 0; j < 4; ++j) blk_bn[b][l][j] = (const float*)d_in[idx++];
      blk_w[b][l] = (const float*)d_in[idx++];
    }
  const float* tr_bn[3][4];
  const float* tr_w[3];
  for (int t = 0; t < 3; ++t) {
    for (int j = 0; j < 4; ++j) tr_bn[t][j] = (const float*)d_in[idx++];
    tr_w[t] = (const float*)d_in[idx++];
  }
  const float* fin_bn[4];
  for (int j = 0; j < 4; ++j) fin_bn[j] = (const float*)d_in[idx++];
  const float* dense_w = (const float*)d_in[idx++];
  const float* dense_b = (const float*)d_in[idx++];

  uint8_t* ws = (uint8_t*)d_ws;
  size_t off = 0;
  auto take = [&](size_t bytes) -> void* {
    void* p = ws + off;
    off = (off + bytes + 255) & ~(size_t)255;
    return p;
  };
  const size_t stageMax = (size_t)Bn * 56 * 56 * 448;
  float*    hbA  = (float*)take(stageMax * 4);
  float*    hbB  = (float*)take(stageMax * 4);
  float*    tmpA = (float*)take(stageMax * 4);
  float*    tmpB = (float*)take((size_t)Bn * 28 * 28 * 576 * 4);
  int8_t*   abin = (int8_t*)take(stageMax);
  int8_t*   wbin = (int8_t*)take((size_t)9 * 64 * 576);
  _Float16* af16 = (_Float16*)take((size_t)Bn * 28 * 28 * 576 * 2);
  _Float16* wf16 = (_Float16*)take((size_t)1000 * 576 * 2);
  float*    gapb = (float*)take((size_t)Bn * 576 * 4);

  {
    long long tot = (long long)Bn * 112 * 112 * 64;
    k_stem_conv<<<cdiv(tot, 256), 256, 0, stream>>>(X, stem_w, hbA, Bn);
    k_bn_act<<<cdiv(tot, 256), 256, 0, stream>>>(
        hbA, stem_bn[0], stem_bn[1], stem_bn[2], stem_bn[3],
        tmpA, Bn * 112 * 112, 64, 64, 64, 1);
    long long pc = (long long)Bn * 56 * 56 * 64;
    k_maxpool<<<cdiv(pc, 256), 256, 0, stream>>>(
        tmpA, hbA, Bn, 112, 112, 56, 56, 64, 64, Cend[0], 3, 2, 0);
  }

  float* cur = hbA;
  float* nxt = hbB;
  for (int b = 0; b < 4; ++b) {
    const int H = Hs[b];
    const int Mpix = Bn * H * H;
    const int pitch = Cend[b];
    for (int l = 0; l < layers[b]; ++l) {
      const int C = Cstart[b] + 64 * l;
      const int Cpad = ((C + 63) / 64) * 64;
      long long bs = (long long)Mpix * Cpad;
      k_bn_sign<<<cdiv(bs, 256), 256, 0, stream>>>(
          cur, blk_bn[b][l][0], blk_bn[b][l][1], blk_bn[b][l][2], blk_bn[b][l][3],
          abin, Mpix, C, pitch, Cpad);
      k_bin_weights<<<cdiv((long long)9 * 64 * Cpad, 256), 256, 0, stream>>>(
          blk_w[b][l], wbin, C, Cpad);
      k_binconv3x3_iu8<<<cdiv(Mpix, 128), 256, 0, stream>>>(
          abin, wbin, cur, Bn, H, H, Cpad, C, pitch);
    }
    if (b < 3) {
      const int C = Cend[b];
      const int Ho = Hs[b + 1];
      const int Mo = Bn * Ho * Ho;
      k_bn_act<<<cdiv((long long)Mpix * C, 256), 256, 0, stream>>>(
          cur, tr_bn[b][0], tr_bn[b][1], tr_bn[b][2], tr_bn[b][3],
          tmpA, Mpix, C, pitch, C, 0);
      k_maxpool<<<cdiv((long long)Mo * C, 256), 256, 0, stream>>>(
          tmpA, tmpB, Bn, H, H, Ho, Ho, C, C, C, 2, 2, 1);
      k_f32_to_f16<<<cdiv((long long)Mo * C, 256), 256, 0, stream>>>(
          tmpB, af16, Mo, C, C);
      k_w_to_f16t<<<cdiv((long long)tcout[b] * C, 256), 256, 0, stream>>>(
          tr_w[b], wf16, C, tcout[b]);
      dim3 g(cdiv(Mo, 128), cdiv(tcout[b], 32));
      k_gemm_f16<<<g, 256, 0, stream>>>(
          af16, wf16, nullptr, nxt, Mo, tcout[b], C, Cend[b + 1], 0);
      float* t = cur; cur = nxt; nxt = t;
    }
  }

  {
    const int C = 576, HW = 49, Mpix = Bn * HW, NC = 1000;
    k_bn_act<<<cdiv((long long)Mpix * C, 256), 256, 0, stream>>>(
        cur, fin_bn[0], fin_bn[1], fin_bn[2], fin_bn[3],
        tmpA, Mpix, C, C, C, 1);
    k_gap<<<cdiv(Bn * C, 256), 256, 0, stream>>>(tmpA, gapb, Bn, HW, C);
    k_f32_to_f16<<<cdiv(Bn * C, 256), 256, 0, stream>>>(gapb, af16, Bn, C, C);
    k_w_to_f16t<<<cdiv((long long)NC * C, 256), 256, 0, stream>>>(
        dense_w, wf16, C, NC);
    dim3 g(cdiv(Bn, 128), cdiv(NC, 32));
    k_gemm_f16<<<g, 256, 0, stream>>>(af16, wf16, dense_b, tmpB, Bn, NC, C, NC, 0);
    k_softmax<<<Bn, 256, 0, stream>>>(tmpB, (float*)d_out, NC);
  }
}
